// LLamaAttention_61211873903128
// MI455X (gfx1250) — matrix operand order repaired and checked by reading, not compiled, not run on hardware
//
#include <hip/hip_runtime.h>
#include <hip/hip_bf16.h>

// ---------------------------------------------------------------------------
// Types for CDNA5 WMMA / TDM (gfx1250, wave32)
// ---------------------------------------------------------------------------
typedef __attribute__((ext_vector_type(8)))  float  v8f;
typedef __attribute__((ext_vector_type(16))) __bf16 v16bf;
typedef __attribute__((ext_vector_type(4)))  unsigned int v4u;
typedef __attribute__((ext_vector_type(4)))  unsigned int u32x4;
typedef __attribute__((ext_vector_type(4)))  int i32x4;
typedef __attribute__((ext_vector_type(8)))  int i32x8;

union Frag16 {
    v4u   u[2];
    v16bf b;
};

__device__ __forceinline__ unsigned short f32_to_bf16(float f) {
    unsigned int u = __float_as_uint(f);
    unsigned int r = u + 0x7FFFu + ((u >> 16) & 1u);   // round-to-nearest-even
    return (unsigned short)(r >> 16);
}

__device__ __forceinline__ v16bf frag_ld2(const unsigned short* p0,
                                          const unsigned short* p1) {
    Frag16 f;
    f.u[0] = *(const v4u*)p0;
    f.u[1] = *(const v4u*)p1;
    return f.b;
}
__device__ __forceinline__ v16bf frag_ld1(const unsigned short* p) {
    Frag16 f;
    f.u[0] = *(const v4u*)p;
    f.u[1] = *(const v4u*)(p + 8);
    return f.b;
}

// ---------------------------------------------------------------------------
// Tensor Data Mover: 2D tile load Global -> LDS (bf16 elements).
// Descriptor per CDNA5 ISA ch.8 (async_tensor): D# group0 + group1, 2D mode.
// LDS row stride = tile_d0 + pad (pad_interval/pad_amount encoded per ISA).
// 6-arg builtin form (amdgpu-toolchain / clang-23).
// ---------------------------------------------------------------------------
__device__ __forceinline__ void tdm_load_2d(unsigned lds_off_bytes,
                                            const void* gptr,
                                            unsigned tile_d0, unsigned tile_d1,
                                            unsigned long long stride0_elems,
                                            unsigned pad_int_enc,
                                            unsigned pad_amt_enc) {
    unsigned long long ga = (unsigned long long)(uintptr_t)gptr;
    u32x4 g0;
    g0.x = 1u;                                  // count=1, user descriptor
    g0.y = lds_off_bytes;                       // lds_addr (bytes)
    g0.z = (unsigned)ga;                        // global_addr[31:0]
    g0.w = (unsigned)((ga >> 32) & 0x01FFFFFFu) | (2u << 30);  // [56:32], type=2
    i32x8 g1;
    // data_size=1 (2B), pad_enable, pad_interval, pad_amount
    g1[0] = (int)((1u << 16) | (1u << 20) | (pad_int_enc << 22) | (pad_amt_enc << 25));
    g1[1] = (int)((tile_d0 & 0xFFFFu) << 16);   // tensor_dim0[15:0] (== tile_d0)
    g1[2] = (int)((tile_d1 & 0xFFFFu) << 16);   // tensor_dim1[15:0] (== tile_d1)
    g1[3] = (int)((tile_d0 & 0xFFFFu) << 16);   // tile_dim0
    g1[4] = (int)(tile_d1 & 0xFFFFu);           // tile_dim1 | tile_dim2=0
    g1[5] = (int)(unsigned)(stride0_elems & 0xFFFFFFFFull);
    g1[6] = (int)(unsigned)((stride0_elems >> 32) & 0xFFFFull);
    g1[7] = 0;
    i32x4 z4 = {0, 0, 0, 0};
    i32x8 z8 = {0, 0, 0, 0, 0, 0, 0, 0};
    __builtin_amdgcn_tensor_load_to_lds(g0, g1, z4, z4, z8, 0);
}

// ---------------------------------------------------------------------------
// Kernel 1: f32 -> bf16 cast (4 elems/thread, b128 loads)
// ---------------------------------------------------------------------------
__global__ __launch_bounds__(256)
void cast_f32_bf16(const float* __restrict__ in, unsigned short* __restrict__ out,
                   int n) {
    int i = (blockIdx.x * 256 + threadIdx.x) * 4;
    if (i < n) {
        float4 v = *(const float4*)(in + i);
        out[i + 0] = f32_to_bf16(v.x);
        out[i + 1] = f32_to_bf16(v.y);
        out[i + 2] = f32_to_bf16(v.z);
        out[i + 3] = f32_to_bf16(v.w);
    }
}

// ---------------------------------------------------------------------------
// Kernel 2: C[M,N] = A_bf16[M,K] @ W_bf16[N,K]^T + bias   (f32 out)
// 128x128 tile / workgroup. TDM double-buffered staging: each wave DMAs its
// own 16-row slab of the A and W tiles for K-step i+1 while computing step i.
// One barrier + one s_wait_tensorcnt per K-step.
// ---------------------------------------------------------------------------
__global__ __launch_bounds__(256)
void gemm_bf16(const unsigned short* __restrict__ A,
               const unsigned short* __restrict__ W,
               const float* __restrict__ bias,
               float* __restrict__ C,
               int M, int N, int K) {
    // LDS row stride 40 halves (80B): tile row 32 halves (16 DW) + 4 DW pad.
    // pad_interval enc 3 -> every 16 DWORDs; pad_amount enc 3 -> 4 DWORDs.
    __shared__ unsigned short As[2][128 * 40];
    __shared__ unsigned short Bs[2][128 * 40];

    const int tid  = threadIdx.x;
    const int lane = tid & 31;
    const int wave = tid >> 5;
    const int m0   = blockIdx.y * 128;
    const int n0   = blockIdx.x * 128;
    const int wm   = (wave & 1) * 64;   // 2 row slabs of 64
    const int wn   = (wave >> 1) * 32;  // 4 col slabs of 32
    const int wslab = wave * 16;        // staging slab (rows) owned by this wave

    const unsigned asb = (unsigned)(uintptr_t)&As[0][0];   // LDS byte offset
    const unsigned bsb = (unsigned)(uintptr_t)&Bs[0][0];
    const unsigned bufBytes = 128 * 40 * 2;

    v8f acc[4][2];
#pragma unroll
    for (int mt = 0; mt < 4; ++mt)
#pragma unroll
        for (int nt = 0; nt < 2; ++nt) acc[mt][nt] = {};

    const int arow  = wm + (lane & 15);
    const int khalf = (lane >> 4) * 8;    // A-frag K sub-chunk (ISA 7.12.2)
    const int bkk   = (lane >> 4) * 8;    // B-frag K sub-chunk (matches A)

    // prologue: DMA first K-step into buffer 0
    tdm_load_2d(asb + wslab * 80, A + (size_t)(m0 + wslab) * K, 32, 16,
                (unsigned long long)K, 3u, 3u);
    tdm_load_2d(bsb + wslab * 80, W + (size_t)(n0 + wslab) * K, 32, 16,
                (unsigned long long)K, 3u, 3u);

    int buf = 0;
    for (int k0 = 0; k0 < K; k0 += 32) {
        __builtin_amdgcn_s_wait_tensorcnt(0);   // own slab landed
        __syncthreads();                        // everyone's slab landed

        int nk = k0 + 32;
        if (nk < K) {                           // prefetch next step (other buffer)
            int nb = buf ^ 1;
            tdm_load_2d(asb + nb * bufBytes + wslab * 80,
                        A + (size_t)(m0 + wslab) * K + nk, 32, 16,
                        (unsigned long long)K, 3u, 3u);
            tdm_load_2d(bsb + nb * bufBytes + wslab * 80,
                        W + (size_t)(n0 + wslab) * K + nk, 32, 16,
                        (unsigned long long)K, 3u, 3u);
        }

        v16bf afr[4], bfr[2];
#pragma unroll
        for (int mt = 0; mt < 4; ++mt) {
            const unsigned short* p = &As[buf][(arow + mt * 16) * 40];
            afr[mt] = frag_ld2(p + khalf, p + 16 + khalf);
        }
#pragma unroll
        for (int nt = 0; nt < 2; ++nt) {
            const unsigned short* p = &Bs[buf][(wn + nt * 16 + (lane & 15)) * 40 + bkk];
            bfr[nt] = frag_ld2(p, p + 16);
        }
#pragma unroll
        for (int mt = 0; mt < 4; ++mt)
#pragma unroll
            for (int nt = 0; nt < 2; ++nt)
                acc[mt][nt] = __builtin_amdgcn_wmma_f32_16x16x32_bf16(
                    false, afr[mt], false, bfr[nt], (short)0, acc[mt][nt],
                    false, false);
        buf ^= 1;
    }

    const int hi = (lane >> 4) << 3;
#pragma unroll
    for (int mt = 0; mt < 4; ++mt)
#pragma unroll
        for (int nt = 0; nt < 2; ++nt) {
            int col = n0 + wn + nt * 16 + (lane & 15);
            float bb = bias[col];
#pragma unroll
            for (int r = 0; r < 8; ++r) {
                int row = m0 + wm + mt * 16 + r + hi;
                C[(size_t)row * N + col] = acc[mt][nt][r] + bb;
            }
        }
}

// ---------------------------------------------------------------------------
// Kernel 3: RoPE (f32 in, bf16 out). One thread per (t, head, d<64) pair.
// ---------------------------------------------------------------------------
__global__ __launch_bounds__(256)
void rope_bf16(const float* __restrict__ in, const int* __restrict__ pos,
               unsigned short* __restrict__ out, int nheads) {
    int idx = blockIdx.x * 256 + threadIdx.x;
    int d = idx & 63;
    int h = (idx >> 6) % nheads;
    int t = idx / (64 * nheads);
    int stride = nheads * 128;
    const float* row = in + (size_t)t * stride + h * 128;
    float x1 = row[d];
    float x2 = row[d + 64];
    float inv = __expf(-0.14391156831212787f * (float)d);  // 10000^(-d/64)
    float ang = (float)pos[t] * inv;
    float s, c;
    __sincosf(ang, &s, &c);
    unsigned short* orow = out + (size_t)t * stride + h * 128;
    orow[d]      = f32_to_bf16(x1 * c - x2 * s);
    orow[d + 64] = f32_to_bf16(x2 * c + x1 * s);
}

// ---------------------------------------------------------------------------
// Kernel 4: flash attention. grid = (T/128 q-blocks, 32 heads), 256 threads.
// K block staged by TDM; V^T staged manually (transpose); probs via LDS.
// ---------------------------------------------------------------------------
__global__ __launch_bounds__(256)
void attn_fa(const unsigned short* __restrict__ Q,    // T x 4096 (rope'd bf16)
             const unsigned short* __restrict__ Kc,   // T x 1024 (rope'd bf16)
             const unsigned short* __restrict__ Vc,   // T x 1024 bf16
             unsigned short* __restrict__ Ctx) {      // T x 4096 bf16
    __shared__ unsigned short Ks [64 * 136];   // K block: [kv_row][dim], 272B stride
    __shared__ unsigned short Vts[128 * 72];   // V block transposed: [dim][kv_row]
    __shared__ unsigned short Ps [128 * 72];   // probs: [q_row][kv_row]

    const int tid  = threadIdx.x;
    const int lane = tid & 31;
    const int wave = tid >> 5;
    const int head = blockIdx.y;
    const int kvh  = head >> 2;
    const int qb   = blockIdx.x;
    const int q0   = qb * 128;
    const int wrow = wave * 16;
    const int hi   = (lane >> 4) << 3;
    const float scale = 0.08838834764831845f;   // 1/sqrt(128)
    const unsigned ksb = (unsigned)(uintptr_t)&Ks[0];

    // Q A-fragments for the whole kernel (dim 128 = 4 K-chunks of 32)
    v16bf qf[4];
    {
        const unsigned short* qp =
            Q + (size_t)(q0 + wrow + (lane & 15)) * 4096 + head * 128;
        int kh = (lane >> 4) * 8;
#pragma unroll
        for (int kc = 0; kc < 4; ++kc)
            qf[kc] = frag_ld2(qp + kc * 32 + kh, qp + kc * 32 + 16 + kh);
    }

    v8f o[8];
#pragma unroll
    for (int dt = 0; dt < 8; ++dt) o[dt] = {};
    float mrow[8], lrow[8];
#pragma unroll
    for (int r = 0; r < 8; ++r) { mrow[r] = -1e30f; lrow[r] = 0.0f; }

    const int jend = 2 * qb + 1;               // causal: kv blocks of 64
    for (int j = 0; j <= jend; ++j) {
        const int k0 = j * 64;
        __syncthreads();                       // prev compute done; safe to overwrite

        // TDM: each wave DMAs 8 K-rows (128 halves/row -> 64 DW; pad 4 DW => 136)
        tdm_load_2d(ksb + wave * 8 * 272,
                    Kc + (size_t)(k0 + wave * 8) * 1024 + kvh * 128,
                    128, 8, 1024ull, 5u, 3u);

        {   // manual transposed staging of V (64x128 -> Vts[dim][kv])
            int r  = tid >> 2;            // kv row 0..63
            int cb = (tid & 3) * 32;      // dim base
            const unsigned short* vp = Vc + (size_t)(k0 + r) * 1024 + kvh * 128 + cb;
            unsigned short tmp[32];
            *(v4u*)&tmp[0]  = *(const v4u*)(vp);
            *(v4u*)&tmp[8]  = *(const v4u*)(vp + 8);
            *(v4u*)&tmp[16] = *(const v4u*)(vp + 16);
            *(v4u*)&tmp[24] = *(const v4u*)(vp + 24);
#pragma unroll
            for (int i2 = 0; i2 < 32; ++i2)
                Vts[(cb + i2) * 72 + r] = tmp[i2];
        }
        __builtin_amdgcn_s_wait_tensorcnt(0);
        __syncthreads();

        // S = Q @ K^T : 4 column tiles of 16 over the 64 kv rows
        v8f s[4];
#pragma unroll
        for (int nt = 0; nt < 4; ++nt) {
            v8f a = {};
            const unsigned short* kb = &Ks[(nt * 16 + (lane & 15)) * 136 + (lane >> 4) * 8];
#pragma unroll
            for (int kc = 0; kc < 4; ++kc) {
                v16bf bf = frag_ld2(kb + kc * 32, kb + kc * 32 + 16);
                a = __builtin_amdgcn_wmma_f32_16x16x32_bf16(
                        false, qf[kc], false, bf, (short)0, a, false, false);
            }
            s[nt] = a;
        }

        // scale + causal mask
#pragma unroll
        for (int nt = 0; nt < 4; ++nt) {
            int kcol = k0 + nt * 16 + (lane & 15);
#pragma unroll
            for (int r = 0; r < 8; ++r) {
                int qrow = q0 + wrow + r + hi;
                float v = s[nt][r] * scale;
                s[nt][r] = (kcol > qrow) ? -1e30f : v;
            }
        }

        // online softmax (row stats across each half-wave's 16 lanes)
        float rmax[8];
#pragma unroll
        for (int r = 0; r < 8; ++r) {
            float m = s[0][r];
#pragma unroll
            for (int nt = 1; nt < 4; ++nt) m = fmaxf(m, s[nt][r]);
            rmax[r] = m;
        }
#pragma unroll
        for (int st = 1; st < 16; st <<= 1)
#pragma unroll
            for (int r = 0; r < 8; ++r)
                rmax[r] = fmaxf(rmax[r], __shfl_xor(rmax[r], st, 32));

        float alpha[8];
#pragma unroll
        for (int r = 0; r < 8; ++r) {
            float mn = fmaxf(mrow[r], rmax[r]);
            alpha[r] = __expf(mrow[r] - mn);
            mrow[r]  = mn;
        }

        float rsum[8];
#pragma unroll
        for (int r = 0; r < 8; ++r) rsum[r] = 0.0f;
#pragma unroll
        for (int nt = 0; nt < 4; ++nt)
#pragma unroll
            for (int r = 0; r < 8; ++r) {
                float p = __expf(s[nt][r] - mrow[r]);
                rsum[r] += p;
                Ps[(wrow + r + hi) * 72 + nt * 16 + (lane & 15)] = f32_to_bf16(p);
            }
#pragma unroll
        for (int st = 1; st < 16; st <<= 1)
#pragma unroll
            for (int r = 0; r < 8; ++r)
                rsum[r] += __shfl_xor(rsum[r], st, 32);
#pragma unroll
        for (int r = 0; r < 8; ++r) lrow[r] = lrow[r] * alpha[r] + rsum[r];

        // O scale + O += P @ V^T  (K dim = 64 kv rows = 2 chunks of 32)
#pragma unroll
        for (int dt = 0; dt < 8; ++dt)
#pragma unroll
            for (int r = 0; r < 8; ++r) o[dt][r] *= alpha[r];

        v16bf pf[2];
        {
            const unsigned short* pp = &Ps[(wrow + (lane & 15)) * 72];
            int kh = (lane >> 4) * 8;
#pragma unroll
            for (int kc = 0; kc < 2; ++kc)
                pf[kc] = frag_ld2(pp + kc * 32 + kh, pp + kc * 32 + 16 + kh);
        }
#pragma unroll
        for (int dt = 0; dt < 8; ++dt) {
            const unsigned short* vb = &Vts[(dt * 16 + (lane & 15)) * 72 + (lane >> 4) * 8];
#pragma unroll
            for (int kc = 0; kc < 2; ++kc) {
                v16bf bf = frag_ld2(vb + kc * 32, vb + kc * 32 + 16);
                o[dt] = __builtin_amdgcn_wmma_f32_16x16x32_bf16(
                            false, pf[kc], false, bf, (short)0, o[dt], false, false);
            }
        }
    }

    // normalize + write context (bf16)
#pragma unroll
    for (int dt = 0; dt < 8; ++dt)
#pragma unroll
        for (int r = 0; r < 8; ++r) {
            int row = q0 + wrow + r + hi;
            int col = dt * 16 + (lane & 15);
            float v = o[dt][r] / lrow[r];
            Ctx[(size_t)row * 4096 + head * 128 + col] = f32_to_bf16(v);
        }
}

// ---------------------------------------------------------------------------
// Host launch. d_ws layout (bytes), total ~184 MB.
// ---------------------------------------------------------------------------
extern "C" void kernel_launch(void* const* d_in, const int* in_sizes, int n_in,
                              void* d_out, int out_size, void* d_ws, size_t ws_size,
                              hipStream_t stream) {
    const float* hs = (const float*)d_in[0];
    const int*   pos = (const int*)d_in[1];
    const float* Wq = (const float*)d_in[2];
    const float* bq = (const float*)d_in[3];
    const float* Wk = (const float*)d_in[4];
    const float* bk = (const float*)d_in[5];
    const float* Wv = (const float*)d_in[6];
    const float* bv = (const float*)d_in[7];
    const float* Wo = (const float*)d_in[8];
    const float* bo = (const float*)d_in[9];
    float* out = (float*)d_out;

    char* ws = (char*)d_ws;
    unsigned short* hs_b  = (unsigned short*)(ws + 0);          // 16 MB
    unsigned short* wq_b  = (unsigned short*)(ws + 16777216);   // 32 MB
    unsigned short* wk_b  = (unsigned short*)(ws + 50331648);   //  8 MB
    unsigned short* wv_b  = (unsigned short*)(ws + 58720256);   //  8 MB
    unsigned short* wo_b  = (unsigned short*)(ws + 67108864);   // 32 MB
    float*          q_f   = (float*)(ws + 100663296);           // 32 MB
    float*          k_f   = (float*)(ws + 134217728);           //  8 MB
    float*          v_f   = (float*)(ws + 142606336);           //  8 MB
    unsigned short* q_b   = (unsigned short*)(ws + 150994944);  // 16 MB
    unsigned short* k_b   = (unsigned short*)(ws + 167772160);  //  4 MB
    unsigned short* v_b   = (unsigned short*)(ws + 171966464);  //  4 MB
    unsigned short* ctx_b = (unsigned short*)(ws + 176160768);  // 16 MB

    // 1) f32 -> bf16 casts
    cast_f32_bf16<<<8388608 / 1024, 256, 0, stream>>>(hs, hs_b, 8388608);
    cast_f32_bf16<<<16777216 / 1024, 256, 0, stream>>>(Wq, wq_b, 16777216);
    cast_f32_bf16<<<4194304 / 1024, 256, 0, stream>>>(Wk, wk_b, 4194304);
    cast_f32_bf16<<<4194304 / 1024, 256, 0, stream>>>(Wv, wv_b, 4194304);
    cast_f32_bf16<<<16777216 / 1024, 256, 0, stream>>>(Wo, wo_b, 16777216);

    // 2) QKV projections (WMMA bf16, f32 out, TDM-staged tiles)
    gemm_bf16<<<dim3(32, 16), 256, 0, stream>>>(hs_b, wq_b, bq, q_f, 2048, 4096, 4096);
    gemm_bf16<<<dim3(8, 16), 256, 0, stream>>>(hs_b, wk_b, bk, k_f, 2048, 1024, 4096);
    gemm_bf16<<<dim3(8, 16), 256, 0, stream>>>(hs_b, wv_b, bv, v_f, 2048, 1024, 4096);

    // 3) RoPE (Q, K) + V cast
    rope_bf16<<<2048 * 32 * 64 / 256, 256, 0, stream>>>(q_f, pos, q_b, 32);
    rope_bf16<<<2048 * 8 * 64 / 256, 256, 0, stream>>>(k_f, pos, k_b, 8);
    cast_f32_bf16<<<2097152 / 1024, 256, 0, stream>>>(v_f, v_b, 2097152);

    // 4) flash attention
    attn_fa<<<dim3(16, 32), 256, 0, stream>>>(q_b, k_b, v_b, ctx_b);

    // 5) output projection -> d_out (f32 + bias)
    gemm_bf16<<<dim3(32, 16), 256, 0, stream>>>(ctx_b, wo_b, bo, out, 2048, 4096, 4096);
}
